// NonMarkovPolicy_47467978556216
// MI455X (gfx1250) — matrix operand order repaired and checked by reading, not compiled, not run on hardware
//
#include <hip/hip_runtime.h>

typedef __attribute__((ext_vector_type(16))) __bf16 v16bf;
typedef __attribute__((ext_vector_type(8)))  float  v8f;
typedef __attribute__((ext_vector_type(4)))  unsigned int u32x4;

#define B_DIM   256
#define S_DIM   512
#define D_DIM   64
#define H_DIM   256
#define FOURH   1024
#define A_DIM   8
#define KTOT    320          // D + H
#define KT_TILES 10          // 320 / 32
#define NT_TILES 64          // 1024 / 16
#define WG_ROWS 16
#define THREADS 512          // 16 waves of 32
#define FRAG_ELEMS 512       // 32 lanes * 16 bf16 per B fragment

__device__ __forceinline__ unsigned short f2bf(float x) {
    unsigned u = __float_as_uint(x);
    unsigned r = (u + 0x7FFFu + ((u >> 16) & 1u)) >> 16;   // RNE
    return (unsigned short)r;
}

__device__ __forceinline__ float sigmoidf(float x) {
    return 1.0f / (1.0f + __expf(-x));
}

// ---------------------------------------------------------------------------
// Pre-pack [Wx; Wh] (fp32, row-major [320][1024]) into bf16 WMMA B-fragments.
// Fragment (kt, nt) holds the 32x16 tile K=[kt*32,kt*32+32), N=[nt*16,nt*16+16)
// laid out lane-major: lane L supplies N = nt*16 + (L&15),
// K_local = (L>>4)*16 + e for e = 0..15 (consecutive bf16 -> consecutive K).
// ---------------------------------------------------------------------------
__global__ void prepack_w_kernel(const float* __restrict__ Wx,
                                 const float* __restrict__ Wh,
                                 unsigned short* __restrict__ wpack) {
    int i = blockIdx.x * blockDim.x + threadIdx.x;
    if (i >= KT_TILES * NT_TILES * FRAG_ELEMS) return;
    int frag   = i >> 9;          // / 512
    int within = i & 511;
    int lane   = within >> 4;
    int e      = within & 15;
    int kt = frag / NT_TILES;
    int nt = frag % NT_TILES;
    int n  = nt * 16 + (lane & 15);
    int k  = kt * 32 + (lane >> 4) * 16 + e;
    float v = (k < D_DIM) ? Wx[k * FOURH + n] : Wh[(k - D_DIM) * FOURH + n];
    wpack[i] = f2bf(v);
}

// ---------------------------------------------------------------------------
// Persistent fused LSTM + heads. One WG per 16 batch rows; 512 sequential
// timesteps. Per step: z = [x_t | h] @ [Wx;Wh] via bf16 WMMA (fp32 acc),
// in-register gates with c carried in VGPRs, h -> LDS, heads -> global.
// B fragments are intentionally re-streamed from the L2-resident packed
// weight buffer every step (see the per-step memory clobber below): keeping
// all 40 fragments live across the loop needs 320 VGPRs/lane and forces
// scratch spills, which is strictly worse than L2 hits.
// ---------------------------------------------------------------------------
__global__ void __launch_bounds__(THREADS)
lstm_policy_kernel(const float* __restrict__ obs,
                   const unsigned short* wpack,          // NOT restrict: may alias 'out'
                   const float* __restrict__ bias,
                   const float* __restrict__ Wm, const float* __restrict__ bm,
                   const float* __restrict__ Ws, const float* __restrict__ bs,
                   float* out) {
    __shared__ unsigned short A_lds[WG_ROWS * KTOT];   // [16][320] bf16 = 10240 B
    __shared__ float h_lds[WG_ROWS * H_DIM];           // [16][256]      = 16384 B
    __shared__ float wm_lds[H_DIM * A_DIM];            //                  8192 B
    __shared__ float ws_lds[H_DIM * A_DIM];            //                  8192 B

    const int tid  = threadIdx.x;
    const int wave = tid >> 5;          // 0..15, owns hidden units [wave*16, wave*16+16)
    const int lane = tid & 31;
    const int r0   = blockIdx.x * WG_ROWS;

    // one-time staging: head weights to LDS, zero the h-region of A
    for (int i = tid; i < H_DIM * A_DIM; i += THREADS) {
        wm_lds[i] = Wm[i];
        ws_lds[i] = Ws[i];
    }
    for (int i = tid; i < WG_ROWS * H_DIM; i += THREADS) {
        int m = i >> 8, j = i & 255;
        A_lds[m * KTOT + D_DIM + j] = 0;               // h_{-1} = 0 (bf16 zero)
    }

    // per-lane constants for the gate stage: this lane's column in each gate block
    const int hid = (wave << 4) + (lane & 15);         // hidden unit index 0..255
    const float bi = bias[0 * H_DIM + hid];
    const float bf = bias[1 * H_DIM + hid];
    const float bg = bias[2 * H_DIM + hid];
    const float bo = bias[3 * H_DIM + hid];

    float creg[8] = {0.f, 0.f, 0.f, 0.f, 0.f, 0.f, 0.f, 0.f};   // cell state, 8 rows/lane

    const int mA    = lane & 15;                // A-fragment row
    const int koffA = (lane >> 4) << 4;         // +16 for upper half-wave

    // base pointer for this wave's B fragments of gate 0 at kt = 0
    const unsigned short* bbase = wpack + (size_t)wave * FRAG_ELEMS + lane * 16;

    __syncthreads();

    #pragma clang loop unroll(disable)
    for (int t = 0; t < S_DIM; ++t) {
        // Optimizer barrier: forbid hoisting the (truly loop-invariant) weight
        // fragment loads out of the timestep loop -> they stay as L2-hitting
        // global_load_b128 instead of 320 VGPRs of spilled scratch traffic.
        asm volatile("" ::: "memory");

        // ---- stage x_t into A_lds[:, 0:64) as bf16 (1024 elems, 2/thread) ----
        {
            int idx = tid * 2;
            int m = idx >> 6;
            int d = idx & 63;
            const float2 xv = *(const float2*)(obs + (size_t)(r0 + m) * (S_DIM * D_DIM)
                                               + (size_t)t * D_DIM + d);
            A_lds[m * KTOT + d]     = f2bf(xv.x);
            A_lds[m * KTOT + d + 1] = f2bf(xv.y);
        }
        __syncthreads();

        // ---- GEMM: this wave computes z for its 16 hidden units in all 4 gates ----
        v8f acc[4] = {};
        #pragma unroll
        for (int kt = 0; kt < KT_TILES; ++kt) {
            // A fragment: contiguous K order (k = 16h + e), one 16-element run in LDS
            v16bf afrag;
            const unsigned short* ap = A_lds + mA * KTOT + kt * 32 + koffA;
            ((u32x4*)&afrag)[0] = *(const u32x4*)(ap);
            ((u32x4*)&afrag)[1] = *(const u32x4*)(ap + 8);
            #pragma unroll
            for (int g = 0; g < 4; ++g) {
                // fragment (kt, ntile = g*16 + wave)
                const unsigned short* bp =
                    bbase + ((size_t)kt * NT_TILES + (size_t)g * 16) * FRAG_ELEMS;
                v16bf bfrag;
                ((u32x4*)&bfrag)[0] = *(const u32x4*)(bp);
                ((u32x4*)&bfrag)[1] = *(const u32x4*)(bp + 8);
                acc[g] = __builtin_amdgcn_wmma_f32_16x16x32_bf16(
                    false, afrag, false, bfrag, (short)0, acc[g], false, false);
            }
        }

        // ---- gates entirely in registers; c stays in VGPRs across steps ----
        {
            const int mbase = (lane >> 4) << 3;   // rows 0..7 vs 8..15
            #pragma unroll
            for (int r = 0; r < 8; ++r) {
                float zi = acc[0][r] + bi;
                float zf = acc[1][r] + bf;
                float zg = acc[2][r] + bg;
                float zo = acc[3][r] + bo;
                float ig = sigmoidf(zi);
                float fg = sigmoidf(zf);
                float gg = tanhf(zg);
                float og = sigmoidf(zo);
                float c  = fg * creg[r] + ig * gg;
                creg[r]  = c;
                float h  = og * tanhf(c);
                int m = r + mbase;
                h_lds[m * H_DIM + hid] = h;                     // for heads
                A_lds[m * KTOT + D_DIM + hid] = f2bf(h);        // A operand for t+1
            }
        }
        __syncthreads();

        // ---- heads: 16 rows x 8 actions x {mean, log_std}, one dot per thread ----
        if (tid < 256) {
            int which = tid >> 7;                  // 0 = mean, 1 = log_std
            int m     = (tid >> 3) & 15;
            int a     = tid & 7;
            const float* W = which ? ws_lds : wm_lds;
            float dot = 0.f;
            #pragma unroll 4
            for (int j = 0; j < H_DIM; ++j)
                dot = __fmaf_rn(h_lds[m * H_DIM + j], W[j * A_DIM + a], dot);
            float v = dot + (which ? bs[a] : bm[a]);
            if (which) v = fminf(fmaxf(v, -10.f), 2.f);
            size_t off = (size_t)which * ((size_t)B_DIM * S_DIM * A_DIM)
                       + ((size_t)(r0 + m) * S_DIM + t) * A_DIM + a;
            out[off] = v;
        }
        // next iteration's x-staging touches only A_lds[:,0:64) (no reader until
        // the next __syncthreads), so no extra barrier needed here.
    }
}

extern "C" void kernel_launch(void* const* d_in, const int* in_sizes, int n_in,
                              void* d_out, int out_size, void* d_ws, size_t ws_size,
                              hipStream_t stream) {
    const float* obs = (const float*)d_in[0];
    const float* Wx  = (const float*)d_in[1];
    const float* Wh  = (const float*)d_in[2];
    const float* b   = (const float*)d_in[3];
    const float* Wm  = (const float*)d_in[4];
    const float* bm  = (const float*)d_in[5];
    const float* Ws  = (const float*)d_in[6];
    const float* bs  = (const float*)d_in[7];
    float* out = (float*)d_out;
    unsigned short* wpack = (unsigned short*)d_ws;   // 320*1024 bf16 = 640 KB

    int total = KT_TILES * NT_TILES * FRAG_ELEMS;    // 327680
    prepack_w_kernel<<<(total + 255) / 256, 256, 0, stream>>>(Wx, Wh, wpack);

    lstm_policy_kernel<<<B_DIM / WG_ROWS, THREADS, 0, stream>>>(
        obs, wpack, b, Wm, bm, Ws, bs, out);
}
